// LowrankGeoFNO_32719060861283
// MI455X (gfx1250) — matrix operand order repaired and checked by reading, not compiled, not run on hardware
//
#include <hip/hip_runtime.h>
#include <math.h>

#define BB 16
#define NN 8192
#define CC 64
#define KK 128
#define RR 4
#define NL 4
#define SPLIT 8

typedef __bf16 bf16;
typedef __bf16 bf16x8 __attribute__((ext_vector_type(8)));
typedef __bf16 bf16x16 __attribute__((ext_vector_type(16)));
typedef float f32x8 __attribute__((ext_vector_type(8)));

union FragAB { bf16x16 v; bf16x8 h[2]; };
union FragO  { bf16x8 v; bf16 e[8]; };

__device__ __forceinline__ f32x8 wmma_bf16(const FragAB& a, const FragAB& b, f32x8 c) {
  // D(16x16,f32) = A(16x32,bf16) x B(32x16,bf16) + C
  return __builtin_amdgcn_wmma_f32_16x16x32_bf16(false, a.v, false, b.v, (short)0, c, false, false);
}

__device__ __forceinline__ float gelu_exact(float v) {
  return 0.5f * v * (1.0f + erff(v * 0.70710678118654752440f));
}

// ---------------------------------------------------------------------------
// Setup: fc0, mask/weight vectors, Fourier bases in both layouts, wconv^T bf16
// ---------------------------------------------------------------------------
__global__ void k_setup(const float* __restrict__ x,
                        const float* __restrict__ fc0_w, const float* __restrict__ fc0_b,
                        const float* __restrict__ wconv,
                        bf16* __restrict__ hT, bf16* __restrict__ hw,
                        bf16* __restrict__ maskv, float* __restrict__ wvec,
                        bf16* __restrict__ bKNc, bf16* __restrict__ bKNs,
                        bf16* __restrict__ bNKc, bf16* __restrict__ bNKs,
                        bf16* __restrict__ wconvT)
{
  int t = blockIdx.x * blockDim.x + threadIdx.x;   // 0 .. BB*NN-1
  if (t < NL * CC * CC) {                          // transpose conv weights (tiny)
    int l = t / (CC * CC); int rem = t % (CC * CC);
    int o = rem / CC; int i = rem % CC;
    wconvT[((size_t)l * CC + o) * CC + i] = (bf16)wconv[((size_t)l * CC + i) * CC + o];
  }
  int b = t / NN, n = t % NN;
  const float* xp = x + (size_t)t * 7;
  float f0 = xp[0], f1 = xp[1], f2 = xp[2];
  float gx = xp[3], gy = xp[4], w = xp[5], m = xp[6];
  float wv = w * (float)NN;                        // weights * size
  wvec[t] = wv;
  maskv[t] = (bf16)m;

  // h = feats @ fc0_w + fc0_b ; store h^T (N,C) and h*wvec in (C,N)
  for (int c = 0; c < CC; ++c) {
    float h = fc0_b[c] + f0 * fc0_w[c] + f1 * fc0_w[CC + c] + f2 * fc0_w[2 * CC + c];
    hT[(size_t)t * CC + c] = (bf16)h;
    hw[((size_t)b * CC + c) * NN + n] = (bf16)(h * wv);
  }

  float px = 6.28318530717958647692f * gx;
  float py = 6.28318530717958647692f * gy;
  for (int k = 0; k < KK; ++k) {
    int kx = k >> 3, ky = k & 7;
    float ang = px * (float)kx + py * (float)ky;
    float sn, cs;
    sincosf(ang, &sn, &cs);
    bf16 cv = (bf16)(cs * m), sv = (bf16)(sn * m);
    bKNc[((size_t)b * KK + k) * NN + n] = cv;
    bKNs[((size_t)b * KK + k) * NN + n] = sv;
    bNKc[(size_t)t * KK + k] = cv;
    bNKs[(size_t)t * KK + k] = sv;
  }
}

// ---------------------------------------------------------------------------
// Forward transform: x_c/x_s[b,i,k] = sum_x hw[b,i,x] * bases[b,k,x]
// one wave per (b, c-tile, k-tile, split); partial f32 tiles per split
// ---------------------------------------------------------------------------
__global__ void k_fwd(const bf16* __restrict__ hw,
                      const bf16* __restrict__ bKNc, const bf16* __restrict__ bKNs,
                      float* __restrict__ xc_part, float* __restrict__ xs_part)
{
  int wid  = (blockIdx.x * blockDim.x + threadIdx.x) >> 5;
  int lane = threadIdx.x & 31;
  int sp = wid & 7, kt = (wid >> 3) & 7, ct = (wid >> 6) & 3, b = wid >> 8;
  int lo = lane & 15, hi = lane >> 4;
  int c0 = ct * 16, k0 = kt * 16;

  const bf16* arow  = hw   + ((size_t)b * CC + c0 + lo) * NN;
  const bf16* brc   = bKNc + ((size_t)b * KK + k0 + lo) * NN;
  const bf16* brs   = bKNs + ((size_t)b * KK + k0 + lo) * NN;

  f32x8 acc_c = {}; f32x8 acc_s = {};
  int xb = sp * (NN / SPLIT), xe = xb + (NN / SPLIT);
  for (int xx = xb; xx < xe; xx += 32) {
    __builtin_prefetch(arow + xx + 256, 0, 1);
    __builtin_prefetch(brc  + xx + 256, 0, 1);
    __builtin_prefetch(brs  + xx + 256, 0, 1);
    FragAB a, bc, bs;
    a.h[0]  = *(const bf16x8*)(arow + xx + hi * 16);
    a.h[1]  = *(const bf16x8*)(arow + xx + hi * 16 + 8);
    bc.h[0] = *(const bf16x8*)(brc + xx + hi * 16);
    bc.h[1] = *(const bf16x8*)(brc + xx + hi * 16 + 8);
    bs.h[0] = *(const bf16x8*)(brs + xx + hi * 16);
    bs.h[1] = *(const bf16x8*)(brs + xx + hi * 16 + 8);
    acc_c = wmma_bf16(a, bc, acc_c);
    acc_s = wmma_bf16(a, bs, acc_s);
  }
  float* oc = xc_part + (((size_t)sp * BB + b) * CC + c0 + hi * 8) * KK + k0 + lo;
  float* os = xs_part + (((size_t)sp * BB + b) * CC + c0 + hi * 8) * KK + k0 + lo;
#pragma unroll
  for (int r = 0; r < 8; ++r) { oc[(size_t)r * KK] = acc_c[r]; os[(size_t)r * KK] = acc_s[r]; }
}

// x_0[b,i] = sum_x hw[b,i,x] * mask[b,x]  (one wave per row)
__global__ void k_x0(const bf16* __restrict__ hw, const bf16* __restrict__ maskv,
                     float* __restrict__ x0)
{
  int wid  = (blockIdx.x * blockDim.x + threadIdx.x) >> 5;  // 0..1023
  int lane = threadIdx.x & 31;
  int b = wid >> 6, i = wid & 63;
  const bf16* row = hw + ((size_t)b * CC + i) * NN;
  const bf16* mk  = maskv + (size_t)b * NN;
  float s = 0.f;
  for (int xx = lane; xx < NN; xx += 32) s += (float)row[xx] * (float)mk[xx];
  for (int off = 16; off; off >>= 1) s += __shfl_down(s, off, 32);
  if (lane == 0) x0[b * CC + i] = s;
}

// ---------------------------------------------------------------------------
// Low-rank mixing per (b,k): reduce split partials, build A_c=2*f_c, A_s=-2*f_s
// ---------------------------------------------------------------------------
__global__ void k_lowrank(int layer,
                          const float* __restrict__ xc_part, const float* __restrict__ xs_part,
                          const float* __restrict__ x0,
                          const float* __restrict__ wc1, const float* __restrict__ wc2,
                          const float* __restrict__ ws1, const float* __restrict__ ws2,
                          const float* __restrict__ w01, const float* __restrict__ w02,
                          bf16* __restrict__ Ac, bf16* __restrict__ As, float* __restrict__ f0buf)
{
  int t = blockIdx.x * blockDim.x + threadIdx.x;
  if (t >= BB * KK) return;
  int b = t / KK, k = t % KK;
  const float inv2 = 1.0f / (float)NN;   // inv_sqrt(N)^2 (applied to w1*w2 product)

  float p1[RR] = {}, p2[RR] = {}, p3[RR] = {}, p4[RR] = {};
  const float* c1 = wc1 + (size_t)layer * CC * RR * KK;
  const float* s1 = ws1 + (size_t)layer * CC * RR * KK;
  for (int i = 0; i < CC; ++i) {
    float xc = 0.f, xs = 0.f;
#pragma unroll
    for (int sp = 0; sp < SPLIT; ++sp) {
      xc += xc_part[(((size_t)sp * BB + b) * CC + i) * KK + k];
      xs += xs_part[(((size_t)sp * BB + b) * CC + i) * KK + k];
    }
    xs = -xs;   // x_s = -einsum(h, wbases_s)
#pragma unroll
    for (int r = 0; r < RR; ++r) {
      float c1v = c1[((size_t)i * RR + r) * KK + k];
      float s1v = s1[((size_t)i * RR + r) * KK + k];
      p1[r] += xc * c1v;  p3[r] += xs * c1v;
      p4[r] += xc * s1v;  p2[r] += xs * s1v;
    }
  }
  const float* c2 = wc2 + (size_t)layer * RR * CC * KK;
  const float* s2 = ws2 + (size_t)layer * RR * CC * KK;
  for (int o = 0; o < CC; ++o) {
    float fc = 0.f, fs = 0.f;
#pragma unroll
    for (int r = 0; r < RR; ++r) {
      float c2v = c2[((size_t)r * CC + o) * KK + k];
      float s2v = s2[((size_t)r * CC + o) * KK + k];
      fc += p1[r] * c2v - p2[r] * s2v;    // f_c = LR(x_c;c) - LR(x_s;s)
      fs += p3[r] * c2v + p4[r] * s2v;    // f_s = LR(x_s;c) + LR(x_c;s)
    }
    Ac[((size_t)b * CC + o) * KK + k] = (bf16)( 2.0f * fc * inv2);
    As[((size_t)b * CC + o) * KK + k] = (bf16)(-2.0f * fs * inv2);
  }
  if (k == 0) {   // zero-mode low-rank
    const float* z1 = w01 + (size_t)layer * CC * RR;
    const float* z2 = w02 + (size_t)layer * RR * CC;
    float p0[RR] = {};
    for (int i = 0; i < CC; ++i) {
      float v = x0[b * CC + i];
#pragma unroll
      for (int r = 0; r < RR; ++r) p0[r] += v * z1[i * RR + r];
    }
    for (int o = 0; o < CC; ++o) {
      float f = 0.f;
#pragma unroll
      for (int r = 0; r < RR; ++r) f += p0[r] * z2[r * CC + o];
      f0buf[b * CC + o] = f * inv2;
    }
  }
}

// ---------------------------------------------------------------------------
// Inverse transform + pointwise conv + bias + GELU, fused. One wave per
// (b, o-tile, x-tile); 10 WMMAs per wave (4 cos + 4 sin + 2 conv).
// ---------------------------------------------------------------------------
__global__ void k_inv(int layer,
                      const bf16* __restrict__ Ac, const bf16* __restrict__ As,
                      const float* __restrict__ f0buf,
                      const bf16* __restrict__ bNKc, const bf16* __restrict__ bNKs,
                      const bf16* __restrict__ hT_old, const bf16* __restrict__ wconvT,
                      const float* __restrict__ bconv,
                      const bf16* __restrict__ maskv, const float* __restrict__ wvec,
                      bf16* __restrict__ hT_new, bf16* __restrict__ hw_new, int apply_gelu)
{
  int wid  = (blockIdx.x * blockDim.x + threadIdx.x) >> 5;
  int lane = threadIdx.x & 31;
  int xt = wid & 511, ot = (wid >> 9) & 3, b = wid >> 11;
  int lo = lane & 15, hi = lane >> 4;
  int x0 = xt * 16, o0 = ot * 16;

  const bf16* acr = Ac   + ((size_t)b * CC + o0 + lo) * KK;
  const bf16* asr = As   + ((size_t)b * CC + o0 + lo) * KK;
  const bf16* bcr = bNKc + ((size_t)b * NN + x0 + lo) * KK;
  const bf16* bsr = bNKs + ((size_t)b * NN + x0 + lo) * KK;

  f32x8 acc = {};
#pragma unroll
  for (int kc = 0; kc < KK; kc += 32) {
    FragAB a, bb;
    a.h[0]  = *(const bf16x8*)(acr + kc + hi * 16);
    a.h[1]  = *(const bf16x8*)(acr + kc + hi * 16 + 8);
    bb.h[0] = *(const bf16x8*)(bcr + kc + hi * 16);
    bb.h[1] = *(const bf16x8*)(bcr + kc + hi * 16 + 8);
    acc = wmma_bf16(a, bb, acc);
    a.h[0]  = *(const bf16x8*)(asr + kc + hi * 16);
    a.h[1]  = *(const bf16x8*)(asr + kc + hi * 16 + 8);
    bb.h[0] = *(const bf16x8*)(bsr + kc + hi * 16);
    bb.h[1] = *(const bf16x8*)(bsr + kc + hi * 16 + 8);
    acc = wmma_bf16(a, bb, acc);
  }
  // pointwise conv: sum_i wconvT[o,i] * h_old[i,x]
  const bf16* wr = wconvT + ((size_t)layer * CC + o0 + lo) * CC;
  const bf16* hr = hT_old + ((size_t)b * NN + x0 + lo) * CC;
#pragma unroll
  for (int ic = 0; ic < CC; ic += 32) {
    FragAB a, bb;
    a.h[0]  = *(const bf16x8*)(wr + ic + hi * 16);
    a.h[1]  = *(const bf16x8*)(wr + ic + hi * 16 + 8);
    bb.h[0] = *(const bf16x8*)(hr + ic + hi * 16);
    bb.h[1] = *(const bf16x8*)(hr + ic + hi * 16 + 8);
    acc = wmma_bf16(a, bb, acc);
  }

  float mval = (float)maskv[(size_t)b * NN + x0 + lo];
  float wv   = wvec[(size_t)b * NN + x0 + lo];
  FragO ho;
#pragma unroll
  for (int r = 0; r < 8; ++r) {
    int o = o0 + 8 * hi + r;
    float v = acc[r] + f0buf[b * CC + o] * mval + bconv[layer * CC + o];
    if (apply_gelu) v = gelu_exact(v);
    ho.e[r] = (bf16)v;
    hw_new[((size_t)b * CC + o) * NN + x0 + lo] = (bf16)(v * wv);
  }
  *(bf16x8*)(hT_new + ((size_t)b * NN + x0 + lo) * CC + o0 + 8 * hi) = ho.v;
}

// ---------------------------------------------------------------------------
// Final MLP: out = gelu(h @ fc1 + b1) @ fc2 + b2   (one thread per point)
// ---------------------------------------------------------------------------
__global__ void k_final(const bf16* __restrict__ hT,
                        const float* __restrict__ fc1_w, const float* __restrict__ fc1_b,
                        const float* __restrict__ fc2_w, const float* __restrict__ fc2_b,
                        float* __restrict__ out)
{
  int t = blockIdx.x * blockDim.x + threadIdx.x;   // BB*NN
  const bf16* hrow = hT + (size_t)t * CC;
  float h[CC];
#pragma unroll
  for (int i = 0; i < CC; ++i) h[i] = (float)hrow[i];
  float o = 0.f;
  for (int j = 0; j < 128; ++j) {
    float s = fc1_b[j];
#pragma unroll
    for (int i = 0; i < CC; ++i) s += h[i] * fc1_w[i * 128 + j];
    o += gelu_exact(s) * fc2_w[j];
  }
  out[t] = o + fc2_b[0];
}

// ---------------------------------------------------------------------------
static inline size_t al256(size_t v) { return (v + 255) & ~(size_t)255; }

extern "C" void kernel_launch(void* const* d_in, const int* in_sizes, int n_in,
                              void* d_out, int out_size, void* d_ws, size_t ws_size,
                              hipStream_t stream) {
  const float* x     = (const float*)d_in[0];
  const float* fc0_w = (const float*)d_in[1];
  const float* fc0_b = (const float*)d_in[2];
  const float* wc1   = (const float*)d_in[3];
  const float* wc2   = (const float*)d_in[4];
  const float* ws1   = (const float*)d_in[5];
  const float* ws2   = (const float*)d_in[6];
  const float* w01   = (const float*)d_in[7];
  const float* w02   = (const float*)d_in[8];
  const float* wconv = (const float*)d_in[9];
  const float* bconv = (const float*)d_in[10];
  const float* fc1_w = (const float*)d_in[11];
  const float* fc1_b = (const float*)d_in[12];
  const float* fc2_w = (const float*)d_in[13];
  const float* fc2_b = (const float*)d_in[14];
  float* out = (float*)d_out;

  char* ws = (char*)d_ws;
  size_t off = 0;
  bf16*  bKNc   = (bf16*)(ws + off); off = al256(off + (size_t)BB * KK * NN * 2);
  bf16*  bKNs   = (bf16*)(ws + off); off = al256(off + (size_t)BB * KK * NN * 2);
  bf16*  bNKc   = (bf16*)(ws + off); off = al256(off + (size_t)BB * NN * KK * 2);
  bf16*  bNKs   = (bf16*)(ws + off); off = al256(off + (size_t)BB * NN * KK * 2);
  bf16*  hT_a   = (bf16*)(ws + off); off = al256(off + (size_t)BB * NN * CC * 2);
  bf16*  hT_b   = (bf16*)(ws + off); off = al256(off + (size_t)BB * NN * CC * 2);
  bf16*  hw     = (bf16*)(ws + off); off = al256(off + (size_t)BB * CC * NN * 2);
  bf16*  maskv  = (bf16*)(ws + off); off = al256(off + (size_t)BB * NN * 2);
  float* wvec   = (float*)(ws + off); off = al256(off + (size_t)BB * NN * 4);
  float* xcpart = (float*)(ws + off); off = al256(off + (size_t)SPLIT * BB * CC * KK * 4);
  float* xspart = (float*)(ws + off); off = al256(off + (size_t)SPLIT * BB * CC * KK * 4);
  float* x0buf  = (float*)(ws + off); off = al256(off + (size_t)BB * CC * 4);
  bf16*  Ac     = (bf16*)(ws + off); off = al256(off + (size_t)BB * CC * KK * 2);
  bf16*  As     = (bf16*)(ws + off); off = al256(off + (size_t)BB * CC * KK * 2);
  float* f0buf  = (float*)(ws + off); off = al256(off + (size_t)BB * CC * 4);
  bf16*  wconvT = (bf16*)(ws + off); off = al256(off + (size_t)NL * CC * CC * 2);
  (void)ws_size; (void)in_sizes; (void)n_in; (void)out_size;

  k_setup<<<(BB * NN) / 256, 256, 0, stream>>>(x, fc0_w, fc0_b, wconv,
                                               hT_a, hw, maskv, wvec,
                                               bKNc, bKNs, bNKc, bNKs, wconvT);

  bf16* hT_cur = hT_a;
  bf16* hT_nxt = hT_b;
  for (int l = 0; l < NL; ++l) {
    // 4096 waves = 512 blocks x 8 waves
    k_fwd<<<512, 256, 0, stream>>>(hw, bKNc, bKNs, xcpart, xspart);
    // 1024 waves
    k_x0<<<128, 256, 0, stream>>>(hw, maskv, x0buf);
    // 2048 threads
    k_lowrank<<<(BB * KK + 255) / 256, 256, 0, stream>>>(l, xcpart, xspart, x0buf,
                                                         wc1, wc2, ws1, ws2, w01, w02,
                                                         Ac, As, f0buf);
    // 32768 waves = 4096 blocks x 8 waves
    k_inv<<<4096, 256, 0, stream>>>(l, Ac, As, f0buf, bNKc, bNKs,
                                    hT_cur, wconvT, bconv, maskv, wvec,
                                    hT_nxt, hw, (l != NL - 1) ? 1 : 0);
    bf16* tmp = hT_cur; hT_cur = hT_nxt; hT_nxt = tmp;
  }

  k_final<<<(BB * NN) / 256, 256, 0, stream>>>(hT_cur, fc1_w, fc1_b, fc2_w, fc2_b, out);
}
